// CustomRNN_7653631722042
// MI455X (gfx1250) — compile-verified, matrix operand order checked by reading, not run on hardware
//
#include <hip/hip_runtime.h>
#include <hip/hip_bf16.h>

typedef __bf16 bf16_t;
typedef __attribute__((ext_vector_type(16))) __bf16 v16bf;
typedef __attribute__((ext_vector_type(8)))  __bf16 v8bf;
typedef __attribute__((ext_vector_type(8)))  float  v8f;

#define T_DIM   2048
#define H_DIM   2048
#define IN_DIM  2048
#define OUT_DIM 2047
#define W2H_LD  4096   // row stride of W_i2h (INPUT_SIZE + HIDDEN)
#define NBLK_REC 64

// ---------------------------------------------------------------------------
// Fragment loader for NT GEMM (both operands K-major).
// CDNA5 16-bit A fragment (16x32): lanes 0-15 hold row M=lane, K = 0..7 then
// 16..23; lanes 16-31 hold row M=lane-16, K = 8..15 then 24..31.  B (32x16)
// mirrors this with N in place of M, so the same loader serves both.
// ---------------------------------------------------------------------------
__device__ __forceinline__ v16bf load_frag_nt(const bf16_t* __restrict__ base,
                                              int ld, int row0, int k0, int lane) {
  const bf16_t* p = base + (size_t)(row0 + (lane & 15)) * ld
                         + (k0 + ((lane >> 4) << 3));
  v8bf lo = *(const v8bf*)(p);        // K = kh .. kh+7
  v8bf hi = *(const v8bf*)(p + 16);   // K = kh+16 .. kh+23
  return __builtin_shufflevector(lo, hi, 0, 1, 2, 3, 4, 5, 6, 7,
                                         8, 9, 10, 11, 12, 13, 14, 15);
}

// ---------------------------------------------------------------------------
// NT GEMM: out[m][n] = sum_k A[m][k]*B[n][k] + bias[n]   (K = 2048)
// Wave tile 32x32 (4 wmma accum frags), block tile 128x64 (8 waves).
// ---------------------------------------------------------------------------
__global__ __launch_bounds__(256)
void gemm_nt_bf16(const bf16_t* __restrict__ A, const bf16_t* __restrict__ B,
                  const float* __restrict__ bias, float* __restrict__ out,
                  int ldc, int nmax) {
  const int lane = threadIdx.x & 31;
  const int wave = threadIdx.x >> 5;
  const int m0 = blockIdx.y * 128 + (wave >> 1) * 32;
  const int n0 = blockIdx.x * 64  + (wave & 1)  * 32;

  v8f c00 = {}, c01 = {}, c10 = {}, c11 = {};

  for (int k0 = 0; k0 < H_DIM; k0 += 32) {
    v16bf a0 = load_frag_nt(A, H_DIM, m0,      k0, lane);
    v16bf a1 = load_frag_nt(A, H_DIM, m0 + 16, k0, lane);
    v16bf b0 = load_frag_nt(B, H_DIM, n0,      k0, lane);
    v16bf b1 = load_frag_nt(B, H_DIM, n0 + 16, k0, lane);
    c00 = __builtin_amdgcn_wmma_f32_16x16x32_bf16(false, a0, false, b0, (short)0, c00, false, false);
    c01 = __builtin_amdgcn_wmma_f32_16x16x32_bf16(false, a0, false, b1, (short)0, c01, false, false);
    c10 = __builtin_amdgcn_wmma_f32_16x16x32_bf16(false, a1, false, b0, (short)0, c10, false, false);
    c11 = __builtin_amdgcn_wmma_f32_16x16x32_bf16(false, a1, false, b1, (short)0, c11, false, false);
  }

  // C/D layout: VGPR v -> M = v (lanes 0-15) or v+8 (lanes 16-31), N = lane&15.
  const int nl = lane & 15;
  const int mb = (lane >> 4) << 3;
  v8f acc[2][2] = { { c00, c01 }, { c10, c11 } };
  for (int i = 0; i < 2; ++i)
    for (int j = 0; j < 2; ++j) {
      int n = n0 + j * 16 + nl;
      if (n >= nmax) continue;
      float bv = bias[n];
#pragma unroll
      for (int v = 0; v < 8; ++v) {
        int m = m0 + i * 16 + mb + v;
        out[(size_t)m * ldc + n] = acc[i][j][v] + bv;
      }
    }
}

// ---------------------------------------------------------------------------
// Build Xc[t][i] = bf16(combined[t][i]) where combined = concat(x, y).T
// (LDS-tiled transpose so both global read and write are coalesced)
// ---------------------------------------------------------------------------
__global__ __launch_bounds__(256)
void pack_input_bf16(const float* __restrict__ x, const float* __restrict__ y,
                     bf16_t* __restrict__ Xc) {
  __shared__ float tile[32][33];
  const int i0 = blockIdx.x * 32;   // input-feature index
  const int t0 = blockIdx.y * 32;   // time index
#pragma unroll
  for (int r = 0; r < 4; ++r) {
    int i = i0 + threadIdx.y + r * 8;
    int t = t0 + threadIdx.x;
    float v = (i < IN_DIM - 1) ? x[(size_t)i * T_DIM + t] : y[t];
    tile[threadIdx.y + r * 8][threadIdx.x] = v;
  }
  __syncthreads();
#pragma unroll
  for (int r = 0; r < 4; ++r) {
    int t = t0 + threadIdx.y + r * 8;
    int i = i0 + threadIdx.x;
    Xc[(size_t)t * IN_DIM + i] = (bf16_t)tile[threadIdx.x][threadIdx.y + r * 8];
  }
}

// W_x = first IN_DIM columns of W_i2h, converted to bf16 (stays K-major)
__global__ __launch_bounds__(256)
void cvt_wx(const float* __restrict__ W_i2h, bf16_t* __restrict__ Wx) {
  size_t idx = (size_t)blockIdx.x * 256 + threadIdx.x;
  int j = (int)(idx >> 11);
  int i = (int)(idx & (IN_DIM - 1));
  Wx[idx] = (bf16_t)W_i2h[(size_t)j * W2H_LD + i];
}

// W_i2o (2047x2048) -> bf16, padded to 2048 rows with zeros
__global__ __launch_bounds__(256)
void cvt_wo(const float* __restrict__ W_i2o, bf16_t* __restrict__ Wo) {
  size_t idx = (size_t)blockIdx.x * 256 + threadIdx.x;
  int n = (int)(idx >> 11);
  int k = (int)(idx & (H_DIM - 1));
  Wo[idx] = (n < OUT_DIM) ? (bf16_t)W_i2o[(size_t)n * H_DIM + k] : (bf16_t)0.0f;
}

__global__ void init_counter(unsigned* c) {
  if (threadIdx.x == 0 && blockIdx.x == 0) *c = 0u;
}

// ---------------------------------------------------------------------------
// Persistent recurrence: h_t = tanh(Z[t] + W_h @ h_{t-1})
// 64 blocks x 256 thr; 32 rows/block, 8 lanes per row (float4 partial dots),
// shfl-tree reduce, device-wide barrier per step (monotonic atomic counter).
// W_h = columns [2048,4096) of W_i2h, read in fp32 (L2-resident, 16 MB).
// ---------------------------------------------------------------------------
__global__ __launch_bounds__(256)
void rnn_recurrence(const float* __restrict__ W_i2h, const float* __restrict__ Z,
                    const float* __restrict__ h0, float* __restrict__ hbuf,
                    bf16_t* __restrict__ Hbf, float* __restrict__ hfinal,
                    unsigned* counter) {
  const int lane  = threadIdx.x & 31;
  const int wave  = threadIdx.x >> 5;
  const int slice = lane & 7;                          // K-slice within the row
  const int row   = blockIdx.x * 32 + wave * 4 + (lane >> 3);
  const float* __restrict__ Wrow = W_i2h + (size_t)row * W2H_LD + IN_DIM;
  float* hb0 = hbuf;
  float* hb1 = hbuf + H_DIM;

  for (int t = 0; t < T_DIM; ++t) {
    const float* hp = (t == 0) ? h0 : ((t & 1) ? hb0 : hb1);
    float sum = 0.0f;
#pragma unroll 8
    for (int j = 0; j < H_DIM / 32; ++j) {
      int k = slice * 4 + j * 32;
      float4 w  = *(const float4*)(Wrow + k);
      float4 h4 = *(const float4*)(hp + k);
      sum += w.x * h4.x + w.y * h4.y + w.z * h4.z + w.w * h4.w;
    }
    sum += __shfl_down(sum, 4, 8);
    sum += __shfl_down(sum, 2, 8);
    sum += __shfl_down(sum, 1, 8);
    if (slice == 0) {
      float hv = tanhf(Z[(size_t)t * H_DIM + row] + sum);
      ((t & 1) ? hb1 : hb0)[row] = hv;
      Hbf[(size_t)t * H_DIM + row] = (bf16_t)hv;
      if (t == T_DIM - 1) hfinal[row] = hv;
    }
    // device-wide barrier (release writes, arrive, spin, acquire)
    __syncthreads();
    if (threadIdx.x == 0) {
      __threadfence();
      atomicAdd(counter, 1u);
      const unsigned target = (unsigned)(t + 1) * gridDim.x;
      while (atomicAdd(counter, 0u) < target) { __builtin_amdgcn_s_sleep(2); }
    }
    __syncthreads();
    __threadfence();
  }
}

// ---------------------------------------------------------------------------
extern "C" void kernel_launch(void* const* d_in, const int* in_sizes, int n_in,
                              void* d_out, int out_size, void* d_ws, size_t ws_size,
                              hipStream_t stream) {
  (void)in_sizes; (void)n_in; (void)out_size; (void)ws_size;
  const float* x      = (const float*)d_in[0];  // (2047, 2048)
  const float* y      = (const float*)d_in[1];  // (1, 2048)
  const float* hidden = (const float*)d_in[2];  // (1, 2048)
  const float* W_i2h  = (const float*)d_in[3];  // (2048, 4096)
  const float* b_i2h  = (const float*)d_in[4];  // (2048,)
  const float* W_i2o  = (const float*)d_in[5];  // (2047, 2048)
  const float* b_i2o  = (const float*)d_in[6];  // (2047,)
  float* out = (float*)d_out;                   // outputs (2048*2047) ++ h_final (2048)

  char* ws = (char*)d_ws;
  float*    Z    = (float*) ws;  ws += (size_t)T_DIM * H_DIM  * sizeof(float);
  bf16_t*   Xc   = (bf16_t*)ws;  ws += (size_t)T_DIM * IN_DIM * sizeof(bf16_t);
  bf16_t*   Wx   = (bf16_t*)ws;  ws += (size_t)H_DIM * IN_DIM * sizeof(bf16_t);
  bf16_t*   Wo   = (bf16_t*)ws;  ws += (size_t)H_DIM * H_DIM  * sizeof(bf16_t);
  bf16_t*   Hbf  = (bf16_t*)ws;  ws += (size_t)T_DIM * H_DIM  * sizeof(bf16_t);
  float*    hbuf = (float*) ws;  ws += (size_t)2 * H_DIM      * sizeof(float);
  unsigned* counter = (unsigned*)ws;

  hipLaunchKernelGGL(init_counter, dim3(1), dim3(64), 0, stream, counter);
  hipLaunchKernelGGL(pack_input_bf16, dim3(64, 64), dim3(32, 8), 0, stream, x, y, Xc);
  hipLaunchKernelGGL(cvt_wx, dim3((H_DIM * IN_DIM) / 256), dim3(256), 0, stream, W_i2h, Wx);
  hipLaunchKernelGGL(cvt_wo, dim3((H_DIM * H_DIM) / 256), dim3(256), 0, stream, W_i2o, Wo);

  // Z = combined @ W_x.T + b_i2h   (WMMA bf16, fp32 accum)
  hipLaunchKernelGGL(gemm_nt_bf16, dim3(32, 16), dim3(256), 0, stream,
                     Xc, Wx, b_i2h, Z, H_DIM, H_DIM);

  // sequential scan: h_t = tanh(Z[t] + W_h @ h_{t-1}); collects Hbf, h_final
  hipLaunchKernelGGL(rnn_recurrence, dim3(NBLK_REC), dim3(256), 0, stream,
                     W_i2h, Z, hidden, hbuf, Hbf, out + (size_t)T_DIM * OUT_DIM, counter);

  // outputs = H @ W_i2o.T + b_i2o  (WMMA bf16, fp32 accum)
  hipLaunchKernelGGL(gemm_nt_bf16, dim3(32, 16), dim3(256), 0, stream,
                     Hbf, Wo, b_i2o, out, OUT_DIM, OUT_DIM);
}
